// MultiHeadCrossAttention_40570261078751
// MI455X (gfx1250) — matrix operand order repaired and checked by reading, not compiled, not run on hardware
//
#include <hip/hip_runtime.h>
#include <hip/hip_bf16.h>

typedef __attribute__((ext_vector_type(16))) _Float16 v16h;
typedef __attribute__((ext_vector_type(8)))  _Float16 v8h;
typedef __attribute__((ext_vector_type(8)))  float    v8f;

#define HEADS 8
#define HDIM  64
#define NQ    1024
#define NKV   2048
#define EMB   512
#define BATCH 8
#define MSZ   512
#define LOG2E 1.4426950408889634f
// mask fill in log2 domain: FP16_MIN * log2(e)  (exp2(fill - m) == exp(FP16_MIN - m/log2e) -> 0)
#define SFILL (-65504.0f * LOG2E)

// ---------------------------------------------------------------------------
// DPP ROW_XMASK butterfly step: lane i reads lane i^MASK within its 16-lane row
// (pure VALU; no LDS, no dscnt waits)
// ---------------------------------------------------------------------------
template<int MASK>
__device__ __forceinline__ float dpp_xor(float v) {
    return __int_as_float(__builtin_amdgcn_update_dpp(
        0, __float_as_int(v), 0x160 | MASK, 0xf, 0xf, true));
}
__device__ __forceinline__ float row_max16(float v) {
    v = fmaxf(v, dpp_xor<1>(v));
    v = fmaxf(v, dpp_xor<2>(v));
    v = fmaxf(v, dpp_xor<4>(v));
    v = fmaxf(v, dpp_xor<8>(v));
    return v;
}
__device__ __forceinline__ float row_sum16(float v) {
    v += dpp_xor<1>(v);
    v += dpp_xor<2>(v);
    v += dpp_xor<4>(v);
    v += dpp_xor<8>(v);
    return v;
}

// ---------------------------------------------------------------------------
// WMMA fragment loads (contiguous k per lane half, same order for A and B, wave32)
// ---------------------------------------------------------------------------
__device__ __forceinline__ v16h cat8(v8h lo, v8h hi) {
    union { struct { v8h a, b; } s; v16h v; } u;
    u.s.a = lo; u.s.b = hi;
    return u.v;
}

// A operand: 16x32 f16 tile. lane m (0..15): row m, K 0..15 contiguous;
// lane m+16: row m, K 16..31
__device__ __forceinline__ v16h load_a_frag(const _Float16* base, int ldk, int lane) {
    int r  = lane & 15;
    int hi = (lane >> 4) << 4;
    const _Float16* p = base + (size_t)r * ldk + hi;
    v8h lo  = *(const v8h*)(p);
    v8h hi8 = *(const v8h*)(p + 8);
    return cat8(lo, hi8);
}

// B operand: 32x16 (K x N) tile fed from row-major N x K (B^T feed).
// lane n (0..15): column n, K k0..k0+15 contiguous; lane n+16: K k0+16..k0+31
__device__ __forceinline__ v16h load_b_frag(const _Float16* base, int ldk, int lane) {
    int c  = lane & 15;
    int hb = (lane >> 4) << 4;
    return *(const v16h*)(base + (size_t)c * ldk + hb);
}

// ---------------------------------------------------------------------------
// f32 -> f16 conversion
// ---------------------------------------------------------------------------
__global__ void cvt_f32_f16(const float* __restrict__ in, _Float16* __restrict__ out, int n) {
    int i = blockIdx.x * blockDim.x + threadIdx.x;
    if (i < n) out[i] = (_Float16)in[i];
}

// ---------------------------------------------------------------------------
// Vt[b][h][d][kv] = KVH[b*NKV + kv][EMB + h*64 + d]
// ---------------------------------------------------------------------------
__global__ void v_transpose(const _Float16* __restrict__ kvh, _Float16* __restrict__ vt) {
    int i = blockIdx.x * blockDim.x + threadIdx.x;
    int kv = i & (NKV - 1);
    int d  = (i >> 11) & (HDIM - 1);
    int h  = (i >> 17) & (HEADS - 1);
    int b  = i >> 20;
    vt[i] = kvh[((size_t)(b * NKV + kv)) * (2 * EMB) + EMB + h * HDIM + d];
}

// ---------------------------------------------------------------------------
// GEMM: C(MxN) = A(MxK,f16) @ B(NxK,f16)^T * scale (+bias).  Wave tile 16x64.
// ---------------------------------------------------------------------------
template<int F16OUT>
__global__ void gemm_nt(const _Float16* __restrict__ A, const _Float16* __restrict__ B,
                        void* __restrict__ C, const float* __restrict__ bias,
                        int M, int N, int K, float scale) {
    const int lane = threadIdx.x & 31;
    const int w    = threadIdx.x >> 5;
    const int m0   = blockIdx.y * 128 + w * 16;
    const int n0   = blockIdx.x * 64;

    v8f acc[4] = {};
    const _Float16* arow = A + (size_t)m0 * K;

    for (int k0 = 0; k0 < K; k0 += 32) {
        v16h af = load_a_frag(arow + k0, K, lane);
#pragma unroll
        for (int t = 0; t < 4; ++t) {
            v16h bf = load_b_frag(B + (size_t)(n0 + t * 16) * K + k0, K, lane);
            acc[t] = __builtin_amdgcn_wmma_f32_16x16x32_f16(
                false, af, false, bf, (short)0, acc[t], false, false);
        }
    }

    const int c  = lane & 15;
    const int rr = (lane >> 4) << 3;
#pragma unroll
    for (int t = 0; t < 4; ++t) {
#pragma unroll
        for (int i = 0; i < 8; ++i) {
            int row = m0 + rr + i;
            int col = n0 + t * 16 + c;
            float v = acc[t][i] * scale;
            if (bias) v += bias[col];
            if (F16OUT) ((_Float16*)C)[(size_t)row * N + col] = (_Float16)v;
            else        ((float*)C)[(size_t)row * N + col]    = v;
        }
    }
}

// ---------------------------------------------------------------------------
// Flash attention.  grid = (NQ/128, H, B), block = 256 (8 waves).
// Each wave owns 16 q rows; streams kv in chunks of 32.  Q carries
// scale*log2(e), so softmax runs in the exp2 domain (bare v_exp_f32).
// ---------------------------------------------------------------------------
__global__ void flash_attn(const _Float16* __restrict__ QH,   // (B*NQ, EMB)
                           const _Float16* __restrict__ KVH,  // (B*NKV, 2*EMB)
                           const _Float16* __restrict__ VT,   // (B,H,64,NKV)
                           const int*      __restrict__ mask, // (B,1,512,512)
                           _Float16*       __restrict__ OH)   // (B*NQ, EMB)
{
    __shared__ _Float16 plds[8][16][32];   // per-wave P staging, 8 KB

    const int lane = threadIdx.x & 31;
    const int w    = threadIdx.x >> 5;
    const int q0   = blockIdx.x * 128 + w * 16;
    const int h    = blockIdx.y;
    const int b    = blockIdx.z;

    const int c  = lane & 15;
    const int rr = (lane >> 4) << 3;       // 0 or 8
    const int hb = (lane >> 4) << 4;       // 0 or 16

    const _Float16* qb  = QH  + (size_t)(b * NQ + q0) * EMB + h * HDIM;
    const _Float16* kb  = KVH + (size_t)(b * NKV) * (2 * EMB) + h * HDIM;
    const _Float16* vtb = VT  + (size_t)((b * HEADS + h) * HDIM) * NKV;
    const int*      mb  = mask + (size_t)b * MSZ * MSZ;

    v16h qf0 = load_a_frag(qb,      EMB, lane);
    v16h qf1 = load_a_frag(qb + 32, EMB, lane);

    v8f   oacc[4] = {};
    float mrow[8], lrow[8];
#pragma unroll
    for (int i = 0; i < 8; ++i) { mrow[i] = -1.0e30f; lrow[i] = 0.0f; }

    const bool qmasked = (q0 >= (NQ - MSZ));

#pragma unroll 1
    for (int kc = 0; kc < NKV / 32; ++kc) {
        const int kv0 = kc * 32;
        const _Float16* kp0 = kb + (size_t)(kv0 + c) * (2 * EMB) + hb;
        const _Float16* vp0 = vtb + (size_t)c * NKV + kv0 + hb;

        // prefetch next chunk's K/V lines (global_prefetch_b8)
        if (kc + 1 < NKV / 32) {
            __builtin_prefetch(kp0 + 32 * (2 * EMB), 0, 1);
            __builtin_prefetch(kp0 + 48 * (2 * EMB), 0, 1);
            __builtin_prefetch(vp0 + 32, 0, 1);
        }

        // ---- S = Q @ K^T  (16 x 32 chunk, two N tiles, d contraction = 64)
        v8f s[2] = {};
#pragma unroll
        for (int t = 0; t < 2; ++t) {
            const _Float16* kp = kp0 + (size_t)(t * 16) * (2 * EMB);
            v16h bk0 = *(const v16h*)(kp);        // d 0..31 slice
            v16h bk1 = *(const v16h*)(kp + 32);   // d 32..63 slice
            s[t] = __builtin_amdgcn_wmma_f32_16x16x32_f16(
                false, qf0, false, bk0, (short)0, s[t], false, false);
            s[t] = __builtin_amdgcn_wmma_f32_16x16x32_f16(
                false, qf1, false, bk1, (short)0, s[t], false, false);
        }

        // ---- mask (last 512 q x last 512 kv corner only; uniform branch)
        if (qmasked && kv0 >= (NKV - MSZ)) {
#pragma unroll
            for (int t = 0; t < 2; ++t) {
                int kvi = kv0 + t * 16 + c - (NKV - MSZ);
#pragma unroll
                for (int i = 0; i < 8; ++i) {
                    int qi = q0 + rr + i - (NQ - MSZ);
                    if (mb[(size_t)qi * MSZ + kvi] == 0) s[t][i] = SFILL;
                }
            }
        }

        // ---- online softmax: DPP max butterfly, per-lane partial sums
#pragma unroll
        for (int i = 0; i < 8; ++i) {
            float cm   = row_max16(fmaxf(s[0][i], s[1][i]));
            float mnew = fmaxf(mrow[i], cm);
            float alpha = exp2f(mrow[i] - mnew);
            mrow[i] = mnew;
            float p0 = exp2f(s[0][i] - mnew);
            float p1 = exp2f(s[1][i] - mnew);
            lrow[i] = lrow[i] * alpha + (p0 + p1);   // reduced once at the end
#pragma unroll
            for (int td = 0; td < 4; ++td) oacc[td][i] *= alpha;
            plds[w][rr + i][c]      = (_Float16)p0;
            plds[w][rr + i][16 + c] = (_Float16)p1;
        }

        // wait for own-wave LDS stores, then re-read P in A-operand layout
        asm volatile("s_wait_dscnt 0" ::: "memory");
        {
            int r  = lane & 15;
            int hi = (lane >> 4) << 4;
            v8h plo = *(const v8h*)&plds[w][r][hi];
            v8h phi = *(const v8h*)&plds[w][r][8 + hi];
            v16h pf = cat8(plo, phi);

            // ---- O += P @ V   (kv contraction = 32, four d tiles)
#pragma unroll
            for (int td = 0; td < 4; ++td) {
                v16h bv = *(const v16h*)(vtb + (size_t)(td * 16 + c) * NKV + kv0 + hb);
                oacc[td] = __builtin_amdgcn_wmma_f32_16x16x32_f16(
                    false, pf, false, bv, (short)0, oacc[td], false, false);
            }
        }
    }

    // ---- finalize: reduce l across the 16-lane row group, O /= l, store
    float linv[8];
#pragma unroll
    for (int i = 0; i < 8; ++i) linv[i] = 1.0f / row_sum16(lrow[i]);
#pragma unroll
    for (int td = 0; td < 4; ++td) {
#pragma unroll
        for (int i = 0; i < 8; ++i) {
            size_t row = (size_t)(b * NQ + q0 + rr + i);
            OH[row * EMB + h * HDIM + td * 16 + c] = (_Float16)(oacc[td][i] * linv[i]);
        }
    }
}

// ---------------------------------------------------------------------------
// launch
// ---------------------------------------------------------------------------
extern "C" void kernel_launch(void* const* d_in, const int* in_sizes, int n_in,
                              void* d_out, int out_size, void* d_ws, size_t ws_size,
                              hipStream_t stream) {
    const float* x_q  = (const float*)d_in[0];   // (8,1024,512)
    const float* x_kv = (const float*)d_in[1];   // (8,2048,512)
    const int*   mask = (const int*)  d_in[2];   // (8,1,512,512)
    const float* Wq   = (const float*)d_in[3];   // (512,512)
    const float* Wkv  = (const float*)d_in[4];   // (1024,512)
    const float* Wo   = (const float*)d_in[5];   // (512,512)
    const float* bo   = (const float*)d_in[6];   // (512,)

    const int nXq  = BATCH * NQ  * EMB;
    const int nXkv = BATCH * NKV * EMB;
    const int nWq  = EMB * EMB;
    const int nWkv = 2 * EMB * EMB;
    const int nWo  = EMB * EMB;
    const int nQ   = nXq;
    const int nKV  = BATCH * NKV * 2 * EMB;
    const int nVt  = BATCH * HEADS * HDIM * NKV;
    const int nAt  = nXq;

    size_t off = 0;
    auto carve = [&](size_t elems) {
        _Float16* p = (_Float16*)((char*)d_ws + off);
        off += elems * sizeof(_Float16);
        return p;
    };
    _Float16* XqH  = carve(nXq);
    _Float16* XkvH = carve(nXkv);
    _Float16* WqH  = carve(nWq);
    _Float16* WkvH = carve(nWkv);
    _Float16* WoH  = carve(nWo);
    _Float16* QH   = carve(nQ);
    _Float16* KVH  = carve(nKV);
    _Float16* VtH  = carve(nVt);
    _Float16* AtH  = carve(nAt);
    (void)ws_size; (void)n_in; (void)in_sizes; (void)out_size;

    dim3 blk(256);
    auto cb = [](int n) { return dim3((n + 255) / 256); };

    cvt_f32_f16<<<cb(nXq),  blk, 0, stream>>>(x_q,  XqH,  nXq);
    cvt_f32_f16<<<cb(nXkv), blk, 0, stream>>>(x_kv, XkvH, nXkv);
    cvt_f32_f16<<<cb(nWq),  blk, 0, stream>>>(Wq,   WqH,  nWq);
    cvt_f32_f16<<<cb(nWkv), blk, 0, stream>>>(Wkv,  WkvH, nWkv);
    cvt_f32_f16<<<cb(nWo),  blk, 0, stream>>>(Wo,   WoH,  nWo);

    // Q = x_q @ Wq^T, scaled by (1/sqrt(64)) * log2(e) for exp2-domain softmax
    gemm_nt<1><<<dim3(EMB / 64, (BATCH * NQ) / 128), blk, 0, stream>>>(
        XqH, WqH, QH, nullptr, BATCH * NQ, EMB, EMB, 0.125f * LOG2E);
    // KV = x_kv @ Wkv^T
    gemm_nt<1><<<dim3((2 * EMB) / 64, (BATCH * NKV) / 128), blk, 0, stream>>>(
        XkvH, WkvH, KVH, nullptr, BATCH * NKV, 2 * EMB, EMB, 1.0f);

    v_transpose<<<cb(nVt), blk, 0, stream>>>(KVH, VtH);

    flash_attn<<<dim3(NQ / 128, HEADS, BATCH), blk, 0, stream>>>(
        QH, KVH, VtH, mask, AtH);

    // out = Attn @ Wo^T + bo   (f32 output)
    gemm_nt<0><<<dim3(EMB / 64, (BATCH * NQ) / 128), blk, 0, stream>>>(
        AtH, WoH, d_out, bo, BATCH * NQ, EMB, EMB, 1.0f);
}
